// MoELayer_1889785610998
// MI455X (gfx1250) — matrix operand order repaired and checked by reading, not compiled, not run on hardware
//
#include <hip/hip_runtime.h>
#include <hip/hip_bf16.h>
#include <math.h>
#include <stdint.h>

typedef __attribute__((ext_vector_type(16))) __bf16 v16bf;
typedef __attribute__((ext_vector_type(8)))  float  v8f;
typedef unsigned int u32x4 __attribute__((ext_vector_type(4)));
typedef int          i32x4 __attribute__((ext_vector_type(4)));
typedef int          i32x8 __attribute__((ext_vector_type(8)));

#define N_TOK 8192
#define DD    1024
#define EE    8
#define FF    4096
#define TOPK  2
#define ROWS  (N_TOK * TOPK)   // 16384 routed rows
#define EPSV  1e-6f
#define AUXW  0.01f

// GEMM tiling: block = 128 rows x 128 cols; 8 waves as 2(row) x 4(col); wave tile 64x32
#define BT_M  128
#define BT_N1 128   // GEMM1 F-tile
#define BT_N2 128   // GEMM2 D-tile
#define BT_K  32
#define LDSTR 40    // bf16 elements per LDS row (40*2=80B keeps 16B alignment)

#if defined(__HIP_DEVICE_COMPILE__) && __has_builtin(__builtin_amdgcn_tensor_load_to_lds)
#define USE_TDM 1
#else
#define USE_TDM 0
#endif

// ---------------- WMMA fragment loaders (wave32, 16x16x32 bf16) ----------------
// A (16x32, MxK): lanes 0-15 -> M=lane, regs hold K{0..7,16..23}; lanes 16-31 -> K{8..15,24..31}
__device__ __forceinline__ v16bf fragA_ld(const __bf16* base, int m0, int lane) {
    int m  = m0 + (lane & 15);
    int ko = (lane >> 4) << 3;           // 0 or 8
    const __bf16* p = base + m * LDSTR + ko;
    v16bf r;
#pragma unroll
    for (int i = 0; i < 8; ++i) r[i] = p[i];
#pragma unroll
    for (int i = 0; i < 8; ++i) r[i + 8] = p[i + 16];
    return r;
}
// B (32x16, KxN) stored in LDS as [n][k]: lanes 0-15 -> N=lane, K{0..7,16..23}; lanes 16-31 -> K{8..15,24..31}
__device__ __forceinline__ v16bf fragB_ld(const __bf16* base, int n0, int lane) {
    int n  = n0 + (lane & 15);
    int kb = (lane >> 4) << 3;           // 0 or 8
    const __bf16* p = base + n * LDSTR + kb;
    v16bf r;
#pragma unroll
    for (int i = 0; i < 8; ++i) r[i] = p[i];
#pragma unroll
    for (int i = 0; i < 8; ++i) r[i + 8] = p[i + 16];
    return r;
}

// ---------------- Kernel 1: RMSNorm + router + top-2 + aux stats ----------------
__global__ __launch_bounds__(256)
void moe_rms_router(const float* __restrict__ x, const float* __restrict__ norm_w,
                    const float* __restrict__ router_w, __bf16* __restrict__ hb,
                    int* __restrict__ topi, float* __restrict__ topw,
                    float* __restrict__ imp_sum, float* __restrict__ load_cnt)
{
    int n = blockIdx.x;
    int tid = threadIdx.x;
    int lane = tid & 31, wave = tid >> 5;
    const float* xr = x + (size_t)n * DD;

    float4 xv = ((const float4*)xr)[tid];          // 256 thr * 4 = 1024
    float ss = xv.x * xv.x + xv.y * xv.y + xv.z * xv.z + xv.w * xv.w;
#pragma unroll
    for (int o = 16; o > 0; o >>= 1) ss += __shfl_down(ss, o, 32);

    __shared__ float sred[8];
    __shared__ float sms;
    if (lane == 0) sred[wave] = ss;
    __syncthreads();
    if (tid == 0) {
        float t = 0.f;
#pragma unroll
        for (int i = 0; i < 8; ++i) t += sred[i];
        sms = t / (float)DD;
    }
    __syncthreads();
    float rs = rsqrtf(sms + EPSV);

    int base = tid * 4;
    float h[4];
#pragma unroll
    for (int i = 0; i < 4; ++i) {
        float xi = (i == 0) ? xv.x : (i == 1) ? xv.y : (i == 2) ? xv.z : xv.w;
        h[i] = xi * rs * norm_w[base + i];
        hb[(size_t)n * DD + base + i] = (__bf16)h[i];
    }

    float part[EE];
#pragma unroll
    for (int e = 0; e < EE; ++e) {
        const float* rw = router_w + (size_t)e * DD + base;
        part[e] = h[0] * rw[0] + h[1] * rw[1] + h[2] * rw[2] + h[3] * rw[3];
    }
#pragma unroll
    for (int o = 16; o > 0; o >>= 1)
#pragma unroll
        for (int e = 0; e < EE; ++e) part[e] += __shfl_down(part[e], o, 32);

    __shared__ float slog[8][EE];
    __shared__ float slg[EE];
    if (lane == 0)
#pragma unroll
        for (int e = 0; e < EE; ++e) slog[wave][e] = part[e];
    __syncthreads();
    if (tid < EE) {
        float t = 0.f;
#pragma unroll
        for (int w = 0; w < 8; ++w) t += slog[w][tid];
        slg[tid] = t;
    }
    __syncthreads();

    if (tid == 0) {
        float mx = slg[0]; int a1 = 0;
#pragma unroll
        for (int e = 1; e < EE; ++e) if (slg[e] > mx) { mx = slg[e]; a1 = e; }
        float se = 0.f, pr[EE];
#pragma unroll
        for (int e = 0; e < EE; ++e) { pr[e] = __expf(slg[e] - mx); se += pr[e]; }
#pragma unroll
        for (int e = 0; e < EE; ++e) atomicAdd(&imp_sum[e], pr[e] / se);
        atomicAdd(&load_cnt[a1], 1.0f);
        float m2 = -INFINITY; int a2 = 0;
#pragma unroll
        for (int e = 0; e < EE; ++e)
            if (e != a1 && slg[e] > m2) { m2 = slg[e]; a2 = e; }
        float p1 = 1.f / (1.f + __expf(m2 - mx));
        topi[n * 2 + 0] = a1;  topi[n * 2 + 1] = a2;
        topw[n * 2 + 0] = p1;  topw[n * 2 + 1] = 1.f - p1;
    }
}

// ---------------- routing compaction ----------------
__global__ void moe_init(int* cnt, int* cur, float* imp, float* load) {
    int t = threadIdx.x;
    if (t < EE) { cnt[t] = 0; cur[t] = 0; imp[t] = 0.f; load[t] = 0.f; }
}
__global__ void moe_count(const int* __restrict__ topi, int* __restrict__ cnt) {
    int i = blockIdx.x * 256 + threadIdx.x;
    if (i < ROWS) atomicAdd(&cnt[topi[i]], 1);
}
__global__ void moe_scan(const int* __restrict__ cnt, int* __restrict__ off, int* __restrict__ cur) {
    if (threadIdx.x == 0) {
        int s = 0;
#pragma unroll
        for (int e = 0; e < EE; ++e) { off[e] = s; cur[e] = s; s += cnt[e]; }
    }
}
__global__ void moe_scatter(const int* __restrict__ topi, const float* __restrict__ topw,
                            int* __restrict__ cur, int* __restrict__ rtok,
                            int* __restrict__ rslot, float* __restrict__ rw) {
    int i = blockIdx.x * 256 + threadIdx.x;
    if (i < ROWS) {
        int e = topi[i];
        int pos = atomicAdd(&cur[e], 1);
        rtok[pos] = i >> 1;
        rslot[pos] = i;
        rw[pos] = topw[i];
    }
}

// ---------------- GEMM1: hdn = silu(h@gate_e) * (h@up_e), bf16 out ----------------
__global__ __launch_bounds__(256)
void moe_gemm1(const __bf16* __restrict__ hb, const float* __restrict__ gate_w,
               const float* __restrict__ up_w, const int* __restrict__ rtok,
               const int* __restrict__ eoff, const int* __restrict__ ecnt,
               __bf16* __restrict__ hdn)
{
    int e = blockIdx.z;
    int cnt = ecnt[e];
    int row0 = blockIdx.y * BT_M;
    if (row0 >= cnt) return;
    int seg = eoff[e];
    int f0 = blockIdx.x * BT_N1;

    __shared__ __bf16 sA[BT_M * LDSTR];
    __shared__ __bf16 sBg[BT_N1 * LDSTR];
    __shared__ __bf16 sBu[BT_N1 * LDSTR];
    __shared__ int sTok[BT_M];

    int tid = threadIdx.x;
    int lane = tid & 31, wave = tid >> 5;
    int wr = wave >> 2, wc = wave & 3;     // 2x4 wave grid; wave tile 64 x 32

    if (tid < BT_M) {
        int r = row0 + tid;
        sTok[tid] = (r < cnt) ? rtok[seg + r] : -1;
    }
    __syncthreads();

    v8f accg[4][2], accu[4][2];
    v8f vz = {};
#pragma unroll
    for (int i = 0; i < 4; ++i)
#pragma unroll
        for (int j = 0; j < 2; ++j) { accg[i][j] = vz; accu[i][j] = vz; }

    const float* gbase = gate_w + (size_t)e * DD * FF;
    const float* ubase = up_w   + (size_t)e * DD * FF;

    for (int k0 = 0; k0 < DD; k0 += BT_K) {
        __syncthreads();
        { // A tile: 128 x 32 bf16 gathered rows; thread -> row=tid>>1, half=tid&1
            int r = tid >> 1, hh = tid & 1;
            __bf16* dst = sA + r * LDSTR + hh * 16;
            int tok = sTok[r];
            if (tok >= 0) {
                const uint4* src = (const uint4*)(hb + (size_t)tok * DD + k0 + hh * 16);
                *(uint4*)dst = src[0];
                *(uint4*)(dst + 8) = src[1];
            } else {
#pragma unroll
                for (int i = 0; i < 16; ++i) dst[i] = (__bf16)0.f;
            }
        }
        { // B tiles: 32 k x 128 f, fp32 -> bf16, stored [f][k]
            int k = tid >> 3, f = (tid & 7) * 16;
            const float* gs = gbase + (size_t)(k0 + k) * FF + f0 + f;
            const float* us = ubase + (size_t)(k0 + k) * FF + f0 + f;
            if (k0 + BT_K < DD) {
                __builtin_prefetch(gs + (size_t)BT_K * FF, 0, 3);
                __builtin_prefetch(us + (size_t)BT_K * FF, 0, 3);
            }
#pragma unroll
            for (int i = 0; i < 16; ++i) {
                sBg[(f + i) * LDSTR + k] = (__bf16)gs[i];
                sBu[(f + i) * LDSTR + k] = (__bf16)us[i];
            }
        }
        __syncthreads();

        v16bf bg[2], bu[2];
#pragma unroll
        for (int nt = 0; nt < 2; ++nt) {
            bg[nt] = fragB_ld(sBg, wc * 32 + nt * 16, lane);
            bu[nt] = fragB_ld(sBu, wc * 32 + nt * 16, lane);
        }
#pragma unroll
        for (int mt = 0; mt < 4; ++mt) {
            v16bf a = fragA_ld(sA, wr * 64 + mt * 16, lane);
#pragma unroll
            for (int nt = 0; nt < 2; ++nt) {
                accg[mt][nt] = __builtin_amdgcn_wmma_f32_16x16x32_bf16(
                    false, a, false, bg[nt], (short)0, accg[mt][nt], false, false);
                accu[mt][nt] = __builtin_amdgcn_wmma_f32_16x16x32_bf16(
                    false, a, false, bu[nt], (short)0, accu[mt][nt], false, false);
            }
        }
    }

    // epilogue: silu(g)*u -> bf16 scattered store
    int nlane = lane & 15;
    int mhalf = (lane >> 4) * 8;
#pragma unroll
    for (int mt = 0; mt < 4; ++mt)
#pragma unroll
        for (int nt = 0; nt < 2; ++nt) {
            int fcol = f0 + wc * 32 + nt * 16 + nlane;
#pragma unroll
            for (int i = 0; i < 8; ++i) {
                int mrel = wr * 64 + mt * 16 + mhalf + i;
                int r = row0 + mrel;
                if (r < cnt) {
                    float g = accg[mt][nt][i];
                    float u = accu[mt][nt][i];
                    float s = (g / (1.f + __expf(-g))) * u;
                    hdn[(size_t)(seg + r) * FF + fcol] = (__bf16)s;
                }
            }
        }
}

// ---------------- GEMM2: out_partial[slot] = w * (hdn_row @ down_e) ----------------
// B tiles (down_w, [F][D] fp32) are DMA'd by the Tensor Data Mover into an LDS
// staging buffer (s_wait_tensorcnt), then converted fp32->bf16 in LDS.
__global__ __launch_bounds__(256)
void moe_gemm2(const __bf16* __restrict__ hdn, const float* __restrict__ down_w,
               const int* __restrict__ rslot, const float* __restrict__ rw,
               const int* __restrict__ eoff, const int* __restrict__ ecnt,
               float* __restrict__ part)
{
    int e = blockIdx.z;
    int cnt = ecnt[e];
    int row0 = blockIdx.y * BT_M;
    if (row0 >= cnt) return;
    int seg = eoff[e];
    int d0 = blockIdx.x * BT_N2;

    __shared__ __bf16 sA[BT_M * LDSTR];
    __shared__ __bf16 sB[BT_N2 * LDSTR];
#if USE_TDM
    __shared__ float sStage[BT_K * BT_N2];   // 32 x 128 fp32 = 16 KB TDM landing zone
#endif
    __shared__ int   sSlot[BT_M];
    __shared__ float sW[BT_M];

    int tid = threadIdx.x;
    int lane = tid & 31, wave = tid >> 5;
    int wr = wave >> 2, wc = wave & 3;       // 2x4 wave grid; wave tile 64 x 32

    if (tid < BT_M) {
        int r = row0 + tid;
        sSlot[tid] = (r < cnt) ? rslot[seg + r] : 0;
        sW[tid]    = (r < cnt) ? rw[seg + r]    : 0.f;
    }
    __syncthreads();

    v8f acc[4][2];
    v8f vz = {};
#pragma unroll
    for (int i = 0; i < 4; ++i)
#pragma unroll
        for (int j = 0; j < 2; ++j) acc[i][j] = vz;

    const float* dbase = down_w + (size_t)e * FF * DD;

    for (int k0 = 0; k0 < FF; k0 += BT_K) {
        __syncthreads();
        { // A tile from hdn (rows contiguous in routed order)
            int r = tid >> 1, hh = tid & 1;
            __bf16* dst = sA + r * LDSTR + hh * 16;
            if (row0 + r < cnt) {
                const uint4* src = (const uint4*)(hdn + (size_t)(seg + row0 + r) * FF + k0 + hh * 16);
                *(uint4*)dst = src[0];
                *(uint4*)(dst + 8) = src[1];
            } else {
#pragma unroll
                for (int i = 0; i < 16; ++i) dst[i] = (__bf16)0.f;
            }
        }
#if USE_TDM
        if (wave == 0) {
            // Tensor DMA descriptor: 2D tile 128(d) x 32(k) of fp32 from down_w[e]
            unsigned long long ga =
                (unsigned long long)(uintptr_t)(dbase + (size_t)k0 * DD + d0);
            u32x4 g0;
            g0[0] = 1u;                                   // count=1 valid descriptor
            g0[1] = (unsigned)(uintptr_t)(&sStage[0]);    // lds_addr
            g0[2] = (unsigned)(ga & 0xffffffffull);       // global_addr[31:0]
            g0[3] = (unsigned)((ga >> 32) & 0x01ffffffull) | (2u << 30); // addr[56:32] | type=2
            i32x8 g1;
            g1[0] = (int)(2u << 16);                      // data_size=4B
            g1[1] = (int)((unsigned)(DD & 0xffff) << 16); // tensor_dim0[15:0] = D
            g1[2] = (int)((unsigned)(FF & 0xffff) << 16); // tensor_dim1[15:0] = F
            g1[3] = (int)((unsigned)(BT_N2) << 16);       // tile_dim0 = 128
            g1[4] = (int)(BT_K);                          // tile_dim1 = 32
            g1[5] = (int)DD;                              // tensor_dim0_stride = D
            g1[6] = 0; g1[7] = 0;
            i32x4 z4 = {0, 0, 0, 0};
#if __clang_major__ >= 23
            i32x8 z8 = {0, 0, 0, 0, 0, 0, 0, 0};
            __builtin_amdgcn_tensor_load_to_lds(g0, g1, z4, z4, z8, 0);
#else
            __builtin_amdgcn_tensor_load_to_lds(g0, g1, z4, z4, 0);
#endif
            __builtin_amdgcn_s_wait_tensorcnt(0);
        }
        __syncthreads();
        { // convert fp32 staging tile -> bf16 sB[d][k]
            int k = tid >> 3, d = (tid & 7) * 16;
#pragma unroll
            for (int i = 0; i < 16; ++i)
                sB[(d + i) * LDSTR + k] = (__bf16)sStage[k * BT_N2 + d + i];
        }
#else
        { // fallback: direct global fp32 -> bf16 sB[d][k]
            int k = tid >> 3, d = (tid & 7) * 16;
            const float* ds = dbase + (size_t)(k0 + k) * DD + d0 + d;
            if (k0 + BT_K < FF) __builtin_prefetch(ds + (size_t)BT_K * DD, 0, 3);
#pragma unroll
            for (int i = 0; i < 16; ++i)
                sB[(d + i) * LDSTR + k] = (__bf16)ds[i];
        }
#endif
        __syncthreads();

        v16bf b[2];
#pragma unroll
        for (int nt = 0; nt < 2; ++nt) b[nt] = fragB_ld(sB, wc * 32 + nt * 16, lane);
#pragma unroll
        for (int mt = 0; mt < 4; ++mt) {
            v16bf a = fragA_ld(sA, wr * 64 + mt * 16, lane);
#pragma unroll
            for (int nt = 0; nt < 2; ++nt)
                acc[mt][nt] = __builtin_amdgcn_wmma_f32_16x16x32_bf16(
                    false, a, false, b[nt], (short)0, acc[mt][nt], false, false);
        }
    }

    int nlane = lane & 15;
    int mhalf = (lane >> 4) * 8;
#pragma unroll
    for (int mt = 0; mt < 4; ++mt)
#pragma unroll
        for (int nt = 0; nt < 2; ++nt) {
            int dcol = d0 + wc * 32 + nt * 16 + nlane;
#pragma unroll
            for (int i = 0; i < 8; ++i) {
                int mrel = wr * 64 + mt * 16 + mhalf + i;
                if (row0 + mrel < cnt)
                    part[(size_t)sSlot[mrel] * DD + dcol] = sW[mrel] * acc[mt][nt][i];
            }
        }
}

// ---------------- combine + aux ----------------
__global__ __launch_bounds__(256)
void moe_combine(const float* __restrict__ x, const float* __restrict__ part,
                 float* __restrict__ out)
{
    size_t i = (size_t)blockIdx.x * 256 + threadIdx.x;
    size_t n = i / DD, d = i - n * DD;
    out[i] = x[i] + part[(n * 2 + 0) * (size_t)DD + d] + part[(n * 2 + 1) * (size_t)DD + d];
}

__global__ void moe_aux(const float* __restrict__ imp, const float* __restrict__ load,
                        float* __restrict__ out_aux)
{
    if (threadIdx.x == 0) {
        float il = 0.f, ll = 0.f;
#pragma unroll
        for (int e = 0; e < EE; ++e) {
            float a = imp[e] / (float)N_TOK * (float)EE;  il += a * a;
            float b = load[e] / (float)N_TOK * (float)EE; ll += b * b;
        }
        out_aux[0] = AUXW * (il / (float)EE + ll / (float)EE);
    }
}

// ---------------- launch ----------------
extern "C" void kernel_launch(void* const* d_in, const int* in_sizes, int n_in,
                              void* d_out, int out_size, void* d_ws, size_t ws_size,
                              hipStream_t stream)
{
    const float* x        = (const float*)d_in[0];
    const float* norm_w   = (const float*)d_in[1];
    const float* router_w = (const float*)d_in[2];
    const float* gate_w   = (const float*)d_in[3];
    const float* up_w     = (const float*)d_in[4];
    const float* down_w   = (const float*)d_in[5];
    float* out = (float*)d_out;

    char* w = (char*)d_ws;
    size_t off = 0;
    auto alloc = [&](size_t bytes) -> void* {
        void* p = w + off;
        off = (off + bytes + 255) & ~(size_t)255;
        return p;
    };

    __bf16* hb    = (__bf16*)alloc((size_t)N_TOK * DD * 2);
    __bf16* hdn   = (__bf16*)alloc((size_t)ROWS * FF * 2);
    float*  part  = (float*) alloc((size_t)ROWS * DD * 4);
    int*    topi  = (int*)   alloc((size_t)ROWS * 4);
    float*  topw  = (float*) alloc((size_t)ROWS * 4);
    int*    rtok  = (int*)   alloc((size_t)ROWS * 4);
    int*    rslot = (int*)   alloc((size_t)ROWS * 4);
    float*  rwbuf = (float*) alloc((size_t)ROWS * 4);
    int*    cnt   = (int*)   alloc(EE * 4);
    int*    eoff  = (int*)   alloc(EE * 4);
    int*    cur   = (int*)   alloc(EE * 4);
    float*  imp   = (float*) alloc(EE * 4);
    float*  load  = (float*) alloc(EE * 4);

    moe_init<<<1, 32, 0, stream>>>(cnt, cur, imp, load);
    moe_rms_router<<<N_TOK, 256, 0, stream>>>(x, norm_w, router_w, hb, topi, topw, imp, load);
    moe_count<<<(ROWS + 255) / 256, 256, 0, stream>>>(topi, cnt);
    moe_scan<<<1, 1, 0, stream>>>(cnt, eoff, cur);
    moe_scatter<<<(ROWS + 255) / 256, 256, 0, stream>>>(topi, topw, cur, rtok, rslot, rwbuf);

    dim3 g1(FF / BT_N1, ROWS / BT_M, EE);   // 32 x 128 x 8, early-exit past segment
    moe_gemm1<<<g1, 256, 0, stream>>>(hb, gate_w, up_w, rtok, eoff, cnt, hdn);

    dim3 g2(DD / BT_N2, ROWS / BT_M, EE);   // 8 x 128 x 8
    moe_gemm2<<<g2, 256, 0, stream>>>(hdn, down_w, rslot, rwbuf, eoff, cnt, part);

    moe_combine<<<(int)(((size_t)N_TOK * DD) / 256), 256, 0, stream>>>(x, part, out);
    moe_aux<<<1, 32, 0, stream>>>(imp, load, out + (size_t)N_TOK * DD);
}
